// DocumentEncoder_8778913153184
// MI455X (gfx1250) — compile-verified, matrix operand order checked by reading, not run on hardware
//
#include <hip/hip_runtime.h>

#define T_LEN 64
#define BATCH 64
#define SENT  512
#define HID   256
#define G3    768     // 3*HID
#define DENC  512     // 2*HID
#define ATTD  256
#define CAT2  1024    // 2*DENC
#define TBR   (T_LEN*BATCH)   // 4096
#define NEGBIG (-1000000.0f)

typedef __attribute__((ext_vector_type(16))) _Float16 v16h;
typedef __attribute__((ext_vector_type(8)))  _Float16 v8h;
typedef __attribute__((ext_vector_type(8)))  float    v8f;

union Frag16 { v16h v; v8h h[2]; };

// Per-lane fragment load for v_wmma_f32_16x16x32_f16.
// Works for both A (row-major M x K) and B (weights stored row-major N x K):
//  lane<16:  row = row0+lane,    K = {k0..k0+7, k0+16..k0+23}
//  lane>=16: row = row0+lane-16, K = {k0+8..k0+15, k0+24..k0+31}
__device__ __forceinline__ v16h load_frag(const _Float16* base, int ld, int row0, int k0, int lane) {
  const int r  = row0 + (lane & 15);
  const int ks = (lane >> 4) * 8;
  Frag16 f;
  f.h[0] = *(const v8h*)(base + (size_t)r * ld + k0 + ks);
  f.h[1] = *(const v8h*)(base + (size_t)r * ld + k0 + ks + 16);
  return f.v;
}

// ---------------- elementwise helpers ----------------

__global__ void __launch_bounds__(256) k_cvt_f16(const float* __restrict__ s, _Float16* __restrict__ d, int n) {
  int i = blockIdx.x * 256 + threadIdx.x;
  if (i < n) d[i] = (_Float16)s[i];
}

// sent_vec (B,T,SENT) f32 -> time-major (T,B,SENT) f16
__global__ void __launch_bounds__(256) k_pack_x(const float* __restrict__ src, _Float16* __restrict__ dst) {
  size_t i = (size_t)blockIdx.x * 256 + threadIdx.x;   // over TBR*SENT
  int c = (int)(i % SENT);
  size_t r = i / SENT;
  int b = (int)(r % BATCH);
  int t = (int)(r / BATCH);
  dst[i] = (_Float16)src[((size_t)b * T_LEN + t) * SENT + c];
}

// dst[t][b] = src[(t<len ? len-1-t : t)][b]   (packed-GRU reverse semantics), f16, width W
__global__ void __launch_bounds__(256) k_reverse_f16(const _Float16* __restrict__ src, _Float16* __restrict__ dst,
                                                     const int* __restrict__ lengths, int W) {
  size_t i = (size_t)blockIdx.x * 256 + threadIdx.x;   // over TBR*W
  int c = (int)(i % W);
  size_t r = i / W;
  int b = (int)(r % BATCH);
  int t = (int)(r / BATCH);
  int len = lengths[b];
  int rt = (t < len) ? (len - 1 - t) : t;
  dst[i] = src[((size_t)rt * BATCH + b) * W + c];
}

__global__ void __launch_bounds__(256) k_mask(const int* __restrict__ lengths, float* __restrict__ m) {
  int i = blockIdx.x * 256 + threadIdx.x;   // over BATCH*T_LEN, mask shape (B,T)
  int t = i % T_LEN;
  int b = i / T_LEN;
  m[i] = (t >= lengths[b]) ? 1.0f : 0.0f;
}

// ---------------- WMMA GEMM: C[M,N] = A[M,K] @ W[N,K]^T (+ bias) ----------------
// 8 waves/block; each wave computes a 16x64 block (1 m-tile x 4 n-tiles), reusing the
// A fragment across 4 WMMAs per k-step. Grid*8 must equal (M/16)*(N/64).

__global__ void __launch_bounds__(256) k_gemm_bias(const _Float16* __restrict__ A,
                                                   const _Float16* __restrict__ W,
                                                   const float* __restrict__ bias,
                                                   float* __restrict__ C,
                                                   int N, int K, int n4) {
  const int lane = threadIdx.x & 31;
  const int wave = threadIdx.x >> 5;
  const int task = blockIdx.x * 8 + wave;
  const int mt  = task / n4;
  const int nt0 = (task % n4) * 4;

  // hint the B tile rows into cache (global_prefetch_b8)
  __builtin_prefetch(W + (size_t)(nt0 * 16 + (lane & 15)) * K, 0, 3);
  __builtin_prefetch(W + (size_t)((nt0 + 2) * 16 + (lane & 15)) * K, 0, 3);

  v8f acc0 = {}, acc1 = {}, acc2 = {}, acc3 = {};
  for (int k = 0; k < K; k += 32) {
    v16h a  = load_frag(A, K, mt * 16, k, lane);
    v16h b0 = load_frag(W, K, (nt0 + 0) * 16, k, lane);
    v16h b1 = load_frag(W, K, (nt0 + 1) * 16, k, lane);
    v16h b2 = load_frag(W, K, (nt0 + 2) * 16, k, lane);
    v16h b3 = load_frag(W, K, (nt0 + 3) * 16, k, lane);
    acc0 = __builtin_amdgcn_wmma_f32_16x16x32_f16(false, a, false, b0, (short)0, acc0, false, false);
    acc1 = __builtin_amdgcn_wmma_f32_16x16x32_f16(false, a, false, b1, (short)0, acc1, false, false);
    acc2 = __builtin_amdgcn_wmma_f32_16x16x32_f16(false, a, false, b2, (short)0, acc2, false, false);
    acc3 = __builtin_amdgcn_wmma_f32_16x16x32_f16(false, a, false, b3, (short)0, acc3, false, false);
  }
  const int row = mt * 16 + (lane >> 4) * 8;
  const int ncl = lane & 15;
#pragma unroll
  for (int j = 0; j < 4; ++j) {
    const v8f acc = (j == 0) ? acc0 : (j == 1) ? acc1 : (j == 2) ? acc2 : acc3;
    const int col = (nt0 + j) * 16 + ncl;
    const float bv = bias ? bias[col] : 0.0f;
#pragma unroll
    for (int i = 0; i < 8; ++i)
      C[(size_t)(row + i) * N + col] = acc[i] + bv;
  }
}

// ---------------- persistent bidirectional GRU recurrence ----------------
// blockIdx.x: 0 = forward, 1 = backward. One 1024-thread (32-wave) workgroup per
// direction, sequential over T. Each step: gh = h16(64x256) @ Whh(768x256)^T via WMMA
// (96 tasks of 1x2 tiles -> 3 tasks/wave, A fragment reused across the n-pair), then
// gates + packed-sequence freeze/zero semantics; h kept f32 in LDS, f16 mirror for the
// next matmul. out is (T,B,DENC); fwd writes cols [0,HID), bwd writes [HID,DENC)
// with re-reversal.

#define GRU_SMEM (BATCH*HID*4 + BATCH*HID*2 + BATCH*G3*4)  // 64K + 32K + 192K = 288K
#define GRU_THREADS 1024

__global__ void __launch_bounds__(GRU_THREADS) k_gru_seq(
    const float* __restrict__ gi_f, const float* __restrict__ gi_b,
    const _Float16* __restrict__ Whh_f, const _Float16* __restrict__ Whh_b,
    const float* __restrict__ bhh_f, const float* __restrict__ bhh_b,
    const int* __restrict__ lengths,
    float* __restrict__ out,
    float* __restrict__ hT_out) {
  const int tid  = threadIdx.x;
  const int lane = tid & 31;
  const int wave = tid >> 5;
  const bool bwd = (blockIdx.x == 1);
  const float*    gi  = bwd ? gi_b  : gi_f;
  const _Float16* Whh = bwd ? Whh_b : Whh_f;
  const float*    bhh = bwd ? bhh_b : bhh_f;

  extern __shared__ char smem[];
  float*    h32 = (float*)smem;                                       // (B,H)
  _Float16* h16 = (_Float16*)(smem + BATCH * HID * 4);                // (B,H)
  float*    gh  = (float*)(smem + BATCH * HID * 4 + BATCH * HID * 2); // (B,G3)

  // pull the recurrence weights toward the caches once
  __builtin_prefetch(Whh + (size_t)tid * (G3 * HID / GRU_THREADS), 0, 3);

  for (int i = tid; i < BATCH * HID; i += GRU_THREADS) { h32[i] = 0.0f; h16[i] = (_Float16)0.0f; }
  __syncthreads();

  for (int t = 0; t < T_LEN; ++t) {
    // gh = h @ Whh^T : M=64 (4 m-tiles), N=768 (48 n-tiles -> 24 pairs), K=256
    for (int tt = wave; tt < 4 * 24; tt += 32) {
      const int mt = tt & 3;
      const int np = tt >> 2;            // n-pair index, nt = 2*np, 2*np+1
      v8f acc0 = {}, acc1 = {};
#pragma unroll
      for (int k = 0; k < HID; k += 32) {
        v16h a  = load_frag(h16, HID, mt * 16, k, lane);
        v16h b0 = load_frag(Whh, HID, (2 * np + 0) * 16, k, lane);
        v16h b1 = load_frag(Whh, HID, (2 * np + 1) * 16, k, lane);
        acc0 = __builtin_amdgcn_wmma_f32_16x16x32_f16(false, a, false, b0, (short)0, acc0, false, false);
        acc1 = __builtin_amdgcn_wmma_f32_16x16x32_f16(false, a, false, b1, (short)0, acc1, false, false);
      }
      const int row = mt * 16 + (lane >> 4) * 8;
      const int c0  = (2 * np + 0) * 16 + (lane & 15);
      const int c1  = (2 * np + 1) * 16 + (lane & 15);
#pragma unroll
      for (int i = 0; i < 8; ++i) {
        gh[(row + i) * G3 + c0] = acc0[i];
        gh[(row + i) * G3 + c1] = acc1[i];
      }
    }
    __syncthreads();

    // gates
    const float* git = gi + (size_t)t * BATCH * G3;
    for (int u = tid; u < BATCH * HID; u += GRU_THREADS) {
      const int b = u / HID;
      const int j = u % HID;
      const int len = lengths[b];
      const float hr = gh[b * G3 + j]           + bhh[j];
      const float hz = gh[b * G3 + HID + j]     + bhh[HID + j];
      const float hn = gh[b * G3 + 2 * HID + j] + bhh[2 * HID + j];
      const float ir  = git[(size_t)b * G3 + j];
      const float iz  = git[(size_t)b * G3 + HID + j];
      const float inn = git[(size_t)b * G3 + 2 * HID + j];
      const float r = 1.0f / (1.0f + __expf(-(ir + hr)));
      const float z = 1.0f / (1.0f + __expf(-(iz + hz)));
      const float n = tanhf(inn + r * hn);
      const float hprev = h32[u];
      const float hnew  = (1.0f - z) * n + z * hprev;
      const bool valid  = (t < len);
      const float hkeep = valid ? hnew : hprev;   // packed: freeze past length
      h32[u] = hkeep;
      h16[u] = (_Float16)hkeep;
      if (!bwd) {
        out[((size_t)t * BATCH + b) * DENC + j] = valid ? hnew : 0.0f;
      } else {
        if (valid) out[((size_t)(len - 1 - t) * BATCH + b) * DENC + HID + j] = hnew;
        else       out[((size_t)t * BATCH + b) * DENC + HID + j] = 0.0f;
      }
    }
    __syncthreads();
  }

  if (hT_out) {
    float* dst = hT_out + (bwd ? BATCH * HID : 0);
    for (int u = tid; u < BATCH * HID; u += GRU_THREADS) dst[u] = h32[u];
  }
}

// ---------------- concat-attention: energy -> softmax -> weighted ctx -> cat(f16) ----------------
// One block per (q,b). energy[t] = sum_a v[a]*tanh(pre[t,b,a]+q[qi,b,a]); masked softmax over t;
// sa = sum_t score[t]*outputs[t,b,:]; cat = [outputs[qi,b,:], sa] stored f16.

__global__ void __launch_bounds__(256) k_attn(const float* __restrict__ pre,      // (T,B,ATTD)
                                              const float* __restrict__ qb,       // (T,B,ATTD)
                                              const float* __restrict__ vvec,     // (ATTD)
                                              const float* __restrict__ outputs,  // (T,B,DENC)
                                              const int* __restrict__ lengths,
                                              _Float16* __restrict__ cat) {       // (T,B,CAT2)
  const int qi = blockIdx.x / BATCH;
  const int b  = blockIdx.x % BATCH;
  const int tid  = threadIdx.x;
  const int lane = tid & 31;
  const int wave = tid >> 5;
  const int len = lengths[b];

  __shared__ float sc[T_LEN];
  __shared__ float s_inv;

  const float* qrow = qb + ((size_t)qi * BATCH + b) * ATTD;
  for (int t = wave; t < T_LEN; t += 8) {
    const float* prow = pre + ((size_t)t * BATCH + b) * ATTD;
    float e = 0.0f;
    for (int a = lane; a < ATTD; a += 32)
      e += vvec[a] * tanhf(prow[a] + qrow[a]);
#pragma unroll
    for (int off = 16; off > 0; off >>= 1) e += __shfl_xor(e, off, 32);
    if (lane == 0) sc[t] = (t < len) ? e : NEGBIG;
  }
  __syncthreads();

  if (tid == 0) {
    float m = sc[0];
    for (int t = 1; t < T_LEN; ++t) m = fmaxf(m, sc[t]);
    float s = 0.0f;
    for (int t = 0; t < T_LEN; ++t) { float ev = __expf(sc[t] - m); sc[t] = ev; s += ev; }
    s_inv = 1.0f / s;
  }
  __syncthreads();

  const float inv = s_inv;
  const size_t catrow = ((size_t)qi * BATCH + b) * CAT2;
  const size_t orow   = ((size_t)qi * BATCH + b) * DENC;
  for (int c = tid; c < DENC; c += 256) {
    float sa = 0.0f;
    for (int t = 0; t < T_LEN; ++t)
      sa += sc[t] * inv * outputs[((size_t)t * BATCH + b) * DENC + c];
    cat[catrow + c]        = (_Float16)outputs[orow + c];
    cat[catrow + DENC + c] = (_Float16)sa;
  }
}

// ---------------- host orchestration ----------------

extern "C" void kernel_launch(void* const* d_in, const int* in_sizes, int n_in,
                              void* d_out, int out_size, void* d_ws, size_t ws_size,
                              hipStream_t stream) {
  const float* sent_vec = (const float*)d_in[0];
  const int*   lengths  = (const int*)  d_in[1];
  const float* Wih_f = (const float*)d_in[2];
  const float* Whh_f = (const float*)d_in[3];
  const float* bih_f = (const float*)d_in[4];
  const float* bhh_f = (const float*)d_in[5];
  const float* Wih_b = (const float*)d_in[6];
  const float* Whh_b = (const float*)d_in[7];
  const float* bih_b = (const float*)d_in[8];
  const float* bhh_b = (const float*)d_in[9];
  const float* Wpre  = (const float*)d_in[10];
  const float* bpre  = (const float*)d_in[11];
  const float* Wq    = (const float*)d_in[12];
  const float* attv  = (const float*)d_in[13];
  const float* saWih_f = (const float*)d_in[14];
  const float* saWhh_f = (const float*)d_in[15];
  const float* sabih_f = (const float*)d_in[16];
  const float* sabhh_f = (const float*)d_in[17];
  const float* saWih_b = (const float*)d_in[18];
  const float* saWhh_b = (const float*)d_in[19];
  const float* sabih_b = (const float*)d_in[20];
  const float* sabhh_b = (const float*)d_in[21];

  float* out = (float*)d_out;
  float* out_hidden = out;                       // (2,B,HID) = 32768
  float* out_newout = out + 2 * BATCH * HID;     // (T,B,DENC) = 2097152
  float* out_mask   = out_newout + (size_t)TBR * DENC;  // (B,T) = 4096

  // workspace carve-up (256B aligned)
  size_t off = 0;
  char* base = (char*)d_ws;
  auto alloc = [&](size_t bytes) -> char* {
    char* p = base + off;
    off += (bytes + 255) & ~(size_t)255;
    return p;
  };
  float*    gi_a   = (float*)alloc((size_t)TBR * G3 * 4);     // gi fwd, reused for GRU2
  float*    gi_b   = (float*)alloc((size_t)TBR * G3 * 4);     // gi bwd, reused for GRU2
  float*    outputs= (float*)alloc((size_t)TBR * DENC * 4);
  float*    pre    = (float*)alloc((size_t)TBR * ATTD * 4);
  float*    qbuf   = (float*)alloc((size_t)TBR * ATTD * 4);
  _Float16* xh_f   = (_Float16*)alloc((size_t)TBR * SENT * 2);
  _Float16* xh_b   = (_Float16*)alloc((size_t)TBR * SENT * 2);
  _Float16* outs_h = (_Float16*)alloc((size_t)TBR * DENC * 2);
  _Float16* cat_h  = (_Float16*)alloc((size_t)TBR * CAT2 * 2);
  _Float16* catr_h = (_Float16*)alloc((size_t)TBR * CAT2 * 2);
  _Float16* hWih_f = (_Float16*)alloc((size_t)G3 * SENT * 2);
  _Float16* hWih_b = (_Float16*)alloc((size_t)G3 * SENT * 2);
  _Float16* hWhh_f = (_Float16*)alloc((size_t)G3 * HID * 2);
  _Float16* hWhh_b = (_Float16*)alloc((size_t)G3 * HID * 2);
  _Float16* hWpre  = (_Float16*)alloc((size_t)ATTD * DENC * 2);
  _Float16* hWq    = (_Float16*)alloc((size_t)ATTD * DENC * 2);
  _Float16* hsaWih_f = (_Float16*)alloc((size_t)G3 * CAT2 * 2);
  _Float16* hsaWih_b = (_Float16*)alloc((size_t)G3 * CAT2 * 2);
  _Float16* hsaWhh_f = (_Float16*)alloc((size_t)G3 * HID * 2);
  _Float16* hsaWhh_b = (_Float16*)alloc((size_t)G3 * HID * 2);
  (void)ws_size; (void)in_sizes; (void)n_in; (void)out_size;

  auto cvt = [&](const float* s, _Float16* d, int n) {
    k_cvt_f16<<<(n + 255) / 256, 256, 0, stream>>>(s, d, n);
  };

  // 1) pack x time-major f16 + reversed copy
  k_pack_x<<<(TBR * SENT) / 256, 256, 0, stream>>>(sent_vec, xh_f);
  k_reverse_f16<<<(TBR * SENT) / 256, 256, 0, stream>>>(xh_f, xh_b, lengths, SENT);

  // 2) weights -> f16
  cvt(Wih_f, hWih_f, G3 * SENT);   cvt(Wih_b, hWih_b, G3 * SENT);
  cvt(Whh_f, hWhh_f, G3 * HID);    cvt(Whh_b, hWhh_b, G3 * HID);
  cvt(Wpre,  hWpre,  ATTD * DENC); cvt(Wq,    hWq,    ATTD * DENC);
  cvt(saWih_f, hsaWih_f, G3 * CAT2); cvt(saWih_b, hsaWih_b, G3 * CAT2);
  cvt(saWhh_f, hsaWhh_f, G3 * HID);  cvt(saWhh_b, hsaWhh_b, G3 * HID);

  // 3) GRU1 input projections: (4096 x 768) = x @ Wih^T + bih ; K=512
  {
    const int n4 = G3 / 64;                             // 12
    const int blocks = (TBR / 16) * n4 / 8;             // 384
    k_gemm_bias<<<blocks, 256, 0, stream>>>(xh_f, hWih_f, bih_f, gi_a, G3, SENT, n4);
    k_gemm_bias<<<blocks, 256, 0, stream>>>(xh_b, hWih_b, bih_b, gi_b, G3, SENT, n4);
  }

  // 4) GRU1 recurrence (both directions), writes outputs + hidden_t
  k_gru_seq<<<2, GRU_THREADS, GRU_SMEM, stream>>>(gi_a, gi_b, hWhh_f, hWhh_b, bhh_f, bhh_b,
                                                  lengths, outputs, out_hidden);

  // 5) attention projections
  cvt(outputs, outs_h, TBR * DENC);
  {
    const int n4 = ATTD / 64;                           // 4
    const int blocks = (TBR / 16) * n4 / 8;             // 128
    k_gemm_bias<<<blocks, 256, 0, stream>>>(outs_h, hWpre, bpre,    pre,  ATTD, DENC, n4);
    k_gemm_bias<<<blocks, 256, 0, stream>>>(outs_h, hWq,   nullptr, qbuf, ATTD, DENC, n4);
  }

  // 6) energy + softmax + weighted context + concat (f16)
  k_attn<<<TBR, 256, 0, stream>>>(pre, qbuf, attv, outputs, lengths, cat_h);
  k_reverse_f16<<<(TBR * CAT2) / 256, 256, 0, stream>>>(cat_h, catr_h, lengths, CAT2);

  // 7) GRU2 input projections: K=1024 (reuse gi buffers)
  {
    const int n4 = G3 / 64;
    const int blocks = (TBR / 16) * n4 / 8;             // 384
    k_gemm_bias<<<blocks, 256, 0, stream>>>(cat_h,  hsaWih_f, sabih_f, gi_a, G3, CAT2, n4);
    k_gemm_bias<<<blocks, 256, 0, stream>>>(catr_h, hsaWih_b, sabih_b, gi_b, G3, CAT2, n4);
  }

  // 8) GRU2 recurrence -> new_outputs directly into d_out
  k_gru_seq<<<2, GRU_THREADS, GRU_SMEM, stream>>>(gi_a, gi_b, hsaWhh_f, hsaWhh_b, sabhh_f, sabhh_b,
                                                  lengths, out_newout, nullptr);

  // 9) mask (B,T)
  k_mask<<<(BATCH * T_LEN) / 256, 256, 0, stream>>>(lengths, out_mask);
}
